// TaskConditionedMoE_41936060678276
// MI455X (gfx1250) — matrix operand order repaired and checked by reading, not compiled, not run on hardware
//
#include <hip/hip_runtime.h>
#include <hip/hip_bf16.h>
#include <math.h>
#include <stdint.h>

typedef __attribute__((ext_vector_type(16))) __bf16 v16bf;
typedef __attribute__((ext_vector_type(2)))  __bf16 v2bf;
typedef __attribute__((ext_vector_type(8)))  float  v8f;

// gcc-style vector types matching the amdgcn builtin prototypes
typedef int      gv4i  __attribute__((vector_size(16)));
typedef unsigned gv4u  __attribute__((vector_size(16)));
typedef int      gv8i  __attribute__((vector_size(32)));
typedef __attribute__((address_space(1))) gv4i* as1_v4i;
typedef __attribute__((address_space(3))) gv4i* as3_v4i;

constexpr int Bc = 2, Tc = 256, Dc = 2048, Ec = 64, Fc = 512, Kc = 4;
constexpr int Nc  = Bc * Tc;        // 512 tokens
constexpr int Hc  = Fc * Kc;        // 2048 shared hidden
constexpr int NSH = 2;
constexpr int BSTR = 40;            // bf16 elems per LDS B-panel column (80B: 16B aligned)
constexpr int HSTR = Fc + 8;        // bf16 elems per LDS h row
constexpr int NPANEL = 192;         // 64 gate + 64 up + 64 down panels (32 K-rows x 512 cols)

#if defined(__has_builtin)
#if __has_builtin(__builtin_amdgcn_tensor_load_to_lds)
#define HAVE_TDM 1
#endif
#if __has_builtin(__builtin_amdgcn_global_load_async_to_lds_b128)
#define HAVE_ASYNC_LDS 1
#endif
#if __has_builtin(__builtin_amdgcn_cvt_pk_bf16_f32)
#define HAVE_CVT_PK 1
#endif
#endif

static __device__ __forceinline__ v2bf cvt2(float a, float b) {
#ifdef HAVE_CVT_PK
  return __builtin_amdgcn_cvt_pk_bf16_f32(a, b);   // v_cvt_pk_bf16_f32
#else
  v2bf r; r[0] = (__bf16)a; r[1] = (__bf16)b; return r;
#endif
}

// ---- panel staging: TDM descriptor DMA (preferred) / per-lane async / sync --
static __device__ __forceinline__ void panel_issue(const float* gsrc, int ld, float* sSt,
                                                   int tid, int wave) {
#if defined(HAVE_TDM)
  if (wave == 0) {                    // one DMA per workgroup; TENSORcnt tracks it
    unsigned lds = (unsigned)(uintptr_t)sSt;
    unsigned long long ga = (unsigned long long)(uintptr_t)gsrc;
    gv4u g0;
    g0[0] = 1u;                                            // count=1, flags=0
    g0[1] = lds;                                           // lds_addr
    g0[2] = (unsigned)ga;                                  // global_addr[31:0]
    g0[3] = (unsigned)((ga >> 32) & 0x1FFFFFFu) | 0x80000000u;  // addr[56:32] | type=2
    gv8i g1;
    g1[0] = 0x00020000;                                    // data_size=4B, no flags
    g1[1] = (int)(((unsigned)ld & 0xFFFFu) << 16);         // tensor_dim0 = ld (row length)
    g1[2] = (int)(32u << 16);                              // tensor_dim1 = 32 rows
    g1[3] = (int)(512u << 16);                             // tile_dim0 = 512
    g1[4] = 32;                                            // tile_dim1 = 32, tile_dim2 = 0
    g1[5] = ld;                                            // tensor_dim0_stride (elements)
    g1[6] = 0;
    g1[7] = 0;                                             // dim1_stride unused (2D)
    gv4i zz = {0, 0, 0, 0};
#if __clang_major__ >= 23
    gv8i z8 = {0, 0, 0, 0, 0, 0, 0, 0};
    __builtin_amdgcn_tensor_load_to_lds(g0, g1, zz, zz, z8, 0);
#else
    __builtin_amdgcn_tensor_load_to_lds(g0, g1, zz, zz, 0);
#endif
  }
#else
#pragma unroll 4
  for (int i = 0; i < 16; ++i) {                 // 4096 x float4 = 32 rows x 512 cols
    int linear = i * 256 + tid;
    int k = linear >> 7, fq = linear & 127;
    const float* g = gsrc + (long)k * ld + fq * 4;
    float* l = sSt + k * 512 + fq * 4;
#ifdef HAVE_ASYNC_LDS
    __builtin_amdgcn_global_load_async_to_lds_b128(
        (as1_v4i)(uintptr_t)g, (as3_v4i)(unsigned)(uintptr_t)l, 0, 0);
#else
    *(float4*)l = *(const float4*)g;
#endif
  }
#endif
}

static __device__ __forceinline__ void panel_sync(int wave) {
#if defined(HAVE_TDM)
  if (wave == 0) {
#if defined(__has_builtin) && __has_builtin(__builtin_amdgcn_s_wait_tensorcnt)
    __builtin_amdgcn_s_wait_tensorcnt(0);
#else
    asm volatile("s_wait_tensorcnt 0" ::: "memory");
#endif
  }
#elif defined(HAVE_ASYNC_LDS)
#if defined(__has_builtin) && __has_builtin(__builtin_amdgcn_s_wait_asynccnt)
  __builtin_amdgcn_s_wait_asynccnt(0);
#else
  asm volatile("s_wait_asynccnt 0" ::: "memory");
#endif
#endif
  __syncthreads();
}

// ---------------- K0: zero output + expert counters -------------------------
__global__ void zero_kernel(float* __restrict__ out, int* __restrict__ cnt, int n) {
  int i = blockIdx.x * 256 + threadIdx.x;
  if (i < n) out[i] = 0.f;
  if (i < Ec) cnt[i] = 0;
}

// ---------------- K1: task conditioning -> dom[B,E] -------------------------
__global__ void task_embed_kernel(const float* __restrict__ x,
                                  const float* __restrict__ te_W1, const float* __restrict__ te_b1,
                                  const float* __restrict__ te_W2, const float* __restrict__ te_b2,
                                  const float* __restrict__ affinity,
                                  float* __restrict__ dom) {
  __shared__ float xm[Dc];
  __shared__ float h1[Dc / 2];
  __shared__ float t4[4];
  int b = blockIdx.x, tid = threadIdx.x;
  for (int d = tid; d < Dc; d += 256) {
    float s = 0.f;
    const float* xp = x + (size_t)b * Tc * Dc + d;
    for (int t = 0; t < Tc; ++t) s += xp[(size_t)t * Dc];
    xm[d] = s * (1.f / Tc);
  }
  __syncthreads();
  for (int o = tid; o < Dc / 2; o += 256) {
    float s = te_b1[o];
    const float* wp = te_W1 + (size_t)o * Dc;
    for (int d = 0; d < Dc; ++d) s += xm[d] * wp[d];
    h1[o] = 0.5f * s * (1.f + erff(s * 0.7071067811865475f));   // exact GELU
  }
  __syncthreads();
  if (tid < 4) {
    float s = te_b2[tid];
    const float* wp = te_W2 + (size_t)tid * (Dc / 2);
    for (int d = 0; d < Dc / 2; ++d) s += h1[d] * wp[d];
    t4[tid] = s;
  }
  __syncthreads();
  if (tid == 0) {
    float m = fmaxf(fmaxf(t4[0], t4[1]), fmaxf(t4[2], t4[3]));
    float e0 = expf(t4[0] - m), e1 = expf(t4[1] - m), e2 = expf(t4[2] - m), e3 = expf(t4[3] - m);
    float inv = 1.f / (e0 + e1 + e2 + e3);
    t4[0] = e0 * inv; t4[1] = e1 * inv; t4[2] = e2 * inv; t4[3] = e3 * inv;
  }
  __syncthreads();
  if (tid < Ec) {
    const float* ap = affinity + (size_t)tid * 4;
    dom[b * Ec + tid] = t4[0] * ap[0] + t4[1] * ap[1] + t4[2] * ap[2] + t4[3] * ap[3];
  }
}

// ---------------- K2: router + top-k + per-expert token gather --------------
__global__ void router_kernel(const float* __restrict__ x, const float* __restrict__ Wr,
                              const float* __restrict__ rbias, const float* __restrict__ dom,
                              int* __restrict__ cnt, int* __restrict__ list,
                              float* __restrict__ wlist) {
  int tid = threadIdx.x;
  int lane = tid & 31;
  int token = blockIdx.x * 8 + (tid >> 5);
  int b = token >> 8;                                  // T == 256
  const float* xr = x + (size_t)token * Dc;
  int e0 = lane * 2, e1 = e0 + 1;
  const float* w0 = Wr + (size_t)e0 * Dc;
  const float* w1 = Wr + (size_t)e1 * Dc;
  float a0 = 0.f, a1 = 0.f;
  for (int d = 0; d < Dc; d += 4) {
    float4 xv = *(const float4*)(xr + d);
    float4 p  = *(const float4*)(w0 + d);
    float4 q  = *(const float4*)(w1 + d);
    a0 += xv.x * p.x + xv.y * p.y + xv.z * p.z + xv.w * p.w;
    a1 += xv.x * q.x + xv.y * q.y + xv.z * q.z + xv.w * q.w;
  }
  float l0 = a0 + 0.5f * dom[b * Ec + e0];
  float l1 = a1 + 0.5f * dom[b * Ec + e1];
  float m = fmaxf(l0, l1);
  for (int off = 16; off; off >>= 1) m = fmaxf(m, __shfl_xor(m, off, 32));
  float p0 = expf(l0 - m), p1 = expf(l1 - m);
  float s = p0 + p1;
  for (int off = 16; off; off >>= 1) s += __shfl_xor(s, off, 32);
  p0 /= s; p1 /= s;
  float v0 = l0 + rbias[e0], v1 = l1 + rbias[e1];
  int sel[4]; float sp[4];
  for (int r = 0; r < 4; ++r) {
    float bv; int bi;
    if (v0 >= v1) { bv = v0; bi = e0; } else { bv = v1; bi = e1; }
    for (int off = 16; off; off >>= 1) {
      float ov = __shfl_xor(bv, off, 32);
      int   oi = __shfl_xor(bi, off, 32);
      if (ov > bv || (ov == bv && oi < bi)) { bv = ov; bi = oi; }
    }
    sel[r] = bi;
    int owner = bi >> 1;
    sp[r] = __shfl((bi & 1) ? p1 : p0, owner, 32);
    if (lane == owner) { if (bi & 1) v1 = -3.4e38f; else v0 = -3.4e38f; }
  }
  float wsum = sp[0] + sp[1] + sp[2] + sp[3] + 1e-8f;
  if (lane < 4) {
    int e = sel[lane];
    float w = sp[lane] / wsum;
    int pos = atomicAdd(cnt + e, 1);
    list[e * Nc + pos]  = token;
    wlist[e * Nc + pos] = w;
  }
}

// ---------------- K3: fused SwiGLU expert GEMMs (bf16 WMMA, TDM staged) -----
// panel idx: [0,128): gate/up pairs (k0 = (idx>>1)*32); [128,192): down panels
static __device__ __forceinline__ void stage_issue(int idx, int tid, int wave,
                                                   const float* gp, const float* up,
                                                   const float* dp, int ldb, float* sSt) {
  if (idx >= NPANEL) return;
  const float* base; int ld; long off0;
  if (idx < 128) {
    base = (idx & 1) ? up : gp;
    ld = ldb;
    off0 = (long)(idx >> 1) * 32 * ldb;
  } else {
    int j = idx - 128;
    int dc = j >> 4, kq = j & 15;
    base = dp; ld = Dc;
    off0 = (long)kq * 32 * Dc + dc * 512;
  }
  panel_issue(base + off0, ld, sSt, tid, wave);
}

static __device__ __forceinline__ void stage_convert(int tid, const float* sSt, __bf16* sB) {
#pragma unroll 4
  for (int i = 0; i < 32; ++i) {                 // 8192 float pairs
    int linear = i * 256 + tid;
    int k = linear >> 8, fp = linear & 255;
    int f = fp * 2;
    float2 v = *(const float2*)(sSt + k * 512 + f);
    v2bf b = cvt2(v.x, v.y);
    sB[(f + 0) * BSTR + k] = b[0];
    sB[(f + 1) * BSTR + k] = b[1];
  }
}

__global__ __launch_bounds__(256) void moe_wmma_kernel(
    const float* __restrict__ x,
    const float* __restrict__ Wg, const float* __restrict__ Wu, const float* __restrict__ Wd,
    const float* __restrict__ Sg, const float* __restrict__ Su, const float* __restrict__ Sd,
    const int* __restrict__ cnt, const int* __restrict__ list,
    const float* __restrict__ wlist, float* __restrict__ out) {
  __shared__ float  sSt[32 * 512];   // 64 KB fp32 staging panel (TDM destination)
  __shared__ __bf16 sB[Fc * BSTR];   // 40 KB bf16 B panel in WMMA fragment order
  __shared__ __bf16 sH[16 * HSTR];   // weighted SwiGLU activations
  __shared__ int    rT[16];
  __shared__ float  rW[16];

  const int tile = blockIdx.x;
  const int pe   = blockIdx.y;
  const int tid  = threadIdx.x;
  const int wave = tid >> 5;
  const int lane = tid & 31;
  const int rsel = lane & 15;
  const int hi   = lane >> 4;

  const float *gp, *up, *dp;
  int ldb, myCnt;
  const bool routed = pe < Ec;
  if (routed) {
    myCnt = cnt[pe];
    if (tile * 16 >= myCnt) return;
    gp = Wg + (size_t)pe * Dc * Fc;
    up = Wu + (size_t)pe * Dc * Fc;
    dp = Wd + (size_t)pe * Fc * Dc;
    ldb = Fc;
  } else {
    int sh = (pe - Ec) >> 2, ch = (pe - Ec) & 3;
    myCnt = Nc;
    gp = Sg + (size_t)sh * Dc * Hc + (size_t)ch * Fc;
    up = Su + (size_t)sh * Dc * Hc + (size_t)ch * Fc;
    dp = Sd + (size_t)sh * Hc * Dc + (size_t)ch * Fc * Dc;
    ldb = Hc;
  }
  stage_issue(0, tid, wave, gp, up, dp, ldb, sSt);   // prime the pipeline (gate k0=0)
  if (tid < 16) {
    int idx = tile * 16 + tid;
    if (routed) {
      bool valid = idx < myCnt;
      rT[tid] = valid ? list[pe * Nc + idx] : 0;
      rW[tid] = valid ? wlist[pe * Nc + idx] : 0.f;
    } else { rT[tid] = idx; rW[tid] = 1.f; }
  }
  __syncthreads();

  const v8f vz = {0.f, 0.f, 0.f, 0.f, 0.f, 0.f, 0.f, 0.f};
  v8f gacc[4] = {vz, vz, vz, vz};
  v8f uacc[4] = {vz, vz, vz, vz};

  const int atok = rT[rsel];
  const float* xrow = x + (size_t)atok * Dc;

  // ---------------- GEMM1: g = x*Wg, u = x*Wu  (K = D = 2048) ---------------
  for (int i = 0; i < 64; ++i) {
    panel_sync(wave);                              // gate panel i landed in sSt
    stage_convert(tid, sSt, sB);
    __syncthreads();
    stage_issue(2 * i + 1, tid, wave, gp, up, dp, ldb, sSt);   // up panel overlaps wmma

    v16bf af;                                      // A fragment from global x
    {
      const float* ap = xrow + i * 32 + hi * 16;
      float4 q0 = *(const float4*)(ap);
      float4 q1 = *(const float4*)(ap + 4);
      float4 q2 = *(const float4*)(ap + 8);
      float4 q3 = *(const float4*)(ap + 12);
      v2bf c0 = cvt2(q0.x, q0.y), c1 = cvt2(q0.z, q0.w);
      v2bf c2 = cvt2(q1.x, q1.y), c3 = cvt2(q1.z, q1.w);
      v2bf c4 = cvt2(q2.x, q2.y), c5 = cvt2(q2.z, q2.w);
      v2bf c6 = cvt2(q3.x, q3.y), c7 = cvt2(q3.z, q3.w);
      af[0] = c0[0]; af[1] = c0[1]; af[2]  = c1[0]; af[3]  = c1[1];
      af[4] = c2[0]; af[5] = c2[1]; af[6]  = c3[0]; af[7]  = c3[1];
      af[8] = c4[0]; af[9] = c4[1]; af[10] = c5[0]; af[11] = c5[1];
      af[12] = c6[0]; af[13] = c6[1]; af[14] = c7[0]; af[15] = c7[1];
    }
#pragma unroll
    for (int t = 0; t < 4; ++t) {
      int fb = (wave * 4 + t) * 16 + rsel;
      v16bf bfr = *(const v16bf*)(sB + fb * BSTR + hi * 16);
      gacc[t] = __builtin_amdgcn_wmma_f32_16x16x32_bf16(false, af, false, bfr,
                                                        (short)0, gacc[t], false, false);
    }

    panel_sync(wave);                              // up panel landed
    stage_convert(tid, sSt, sB);
    __syncthreads();
    stage_issue(2 * i + 2, tid, wave, gp, up, dp, ldb, sSt);   // next gate (or down 0)
#pragma unroll
    for (int t = 0; t < 4; ++t) {
      int fb = (wave * 4 + t) * 16 + rsel;
      v16bf bfr = *(const v16bf*)(sB + fb * BSTR + hi * 16);
      uacc[t] = __builtin_amdgcn_wmma_f32_16x16x32_bf16(false, af, false, bfr,
                                                        (short)0, uacc[t], false, false);
    }
  }

  // h = silu(g)*u, weighted by combine; park in LDS (down panel 0 is in flight)
#pragma unroll
  for (int t = 0; t < 4; ++t) {
    int cb = (wave * 4 + t) * 16 + rsel;
#pragma unroll
    for (int j = 0; j < 8; j += 2) {
      int r = j + hi * 8;
      float g0 = gacc[t][j],     u0 = uacc[t][j];
      float g1 = gacc[t][j + 1], u1 = uacc[t][j + 1];
      float h0 = (g0 / (1.f + expf(-g0))) * u0 * rW[r];
      float h1 = (g1 / (1.f + expf(-g1))) * u1 * rW[r + 1];
      v2bf b = cvt2(h0, h1);
      sH[(r + 0) * HSTR + cb] = b[0];
      sH[(r + 1) * HSTR + cb] = b[1];
    }
  }
  __syncthreads();

  // ---------------- GEMM2: out += h * Wd  (K = 512, N = 2048 in 4 chunks) ---
  v8f acc[4] = {vz, vz, vz, vz};
  for (int j = 0; j < 64; ++j) {                   // flat (dc, k) panel loop
    panel_sync(wave);                              // down panel j landed
    stage_convert(tid, sSt, sB);
    __syncthreads();
    stage_issue(129 + j, tid, wave, gp, up, dp, ldb, sSt);

    v16bf af;
    {
      const __bf16* hp = sH + rsel * HSTR + (j & 15) * 32 + hi * 16;
#pragma unroll
      for (int q = 0; q < 8; ++q) { af[q] = hp[q]; af[8 + q] = hp[8 + q]; }
    }
#pragma unroll
    for (int t = 0; t < 4; ++t) {
      int fb = (wave * 4 + t) * 16 + rsel;
      v16bf bfr = *(const v16bf*)(sB + fb * BSTR + hi * 16);
      acc[t] = __builtin_amdgcn_wmma_f32_16x16x32_bf16(false, af, false, bfr,
                                                       (short)0, acc[t], false, false);
    }
    if ((j & 15) == 15) {                          // finished one 512-col output chunk
      int dc = j >> 4;
#pragma unroll
      for (int t = 0; t < 4; ++t) {
        int colg = dc * 512 + (wave * 4 + t) * 16 + rsel;
#pragma unroll
        for (int q = 0; q < 8; ++q) {
          int r = q + hi * 8;
          if (!routed || (tile * 16 + r) < myCnt) {
            atomicAdd(out + (size_t)rT[r] * Dc + colg, acc[t][q]);
          }
        }
        acc[t] = vz;
      }
    }
  }
}

// ---------------------------------------------------------------------------
extern "C" void kernel_launch(void* const* d_in, const int* in_sizes, int n_in,
                              void* d_out, int out_size, void* d_ws, size_t ws_size,
                              hipStream_t stream) {
  (void)in_sizes; (void)n_in; (void)out_size; (void)ws_size;
  const float* x        = (const float*)d_in[0];
  const float* Wr       = (const float*)d_in[1];
  const float* rbias    = (const float*)d_in[2];
  const float* affinity = (const float*)d_in[3];
  const float* te_W1    = (const float*)d_in[4];
  const float* te_b1    = (const float*)d_in[5];
  const float* te_W2    = (const float*)d_in[6];
  const float* te_b2    = (const float*)d_in[7];
  const float* Wg       = (const float*)d_in[8];
  const float* Wu       = (const float*)d_in[9];
  const float* Wd       = (const float*)d_in[10];
  const float* Sg       = (const float*)d_in[11];
  const float* Su       = (const float*)d_in[12];
  const float* Sd       = (const float*)d_in[13];
  float* out = (float*)d_out;

  char*  ws    = (char*)d_ws;
  float* dom   = (float*)(ws);
  int*   cnt   = (int*)(ws + 1024);
  int*   list  = (int*)(ws + 2048);
  float* wlist = (float*)(ws + 2048 + (size_t)Ec * Nc * 4);

  int ntot = Nc * Dc;
  zero_kernel<<<(ntot + 255) / 256, 256, 0, stream>>>(out, cnt, ntot);
  task_embed_kernel<<<Bc, 256, 0, stream>>>(x, te_W1, te_b1, te_W2, te_b2, affinity, dom);
  router_kernel<<<Nc / 8, 256, 0, stream>>>(x, Wr, rbias, dom, cnt, list, wlist);
  dim3 grid(Nc / 16, Ec + 4 * NSH);
  moe_wmma_kernel<<<grid, 256, 0, stream>>>(x, Wg, Wu, Wd, Sg, Su, Sd, cnt, list, wlist, out);
}
